// CollectAtomTriples_86002425135351
// MI455X (gfx1250) — matrix operand order repaired and checked by reading, not compiled, not run on hardware
//
#include <hip/hip_runtime.h>
#include <hip/hip_bf16.h>

typedef __attribute__((ext_vector_type(16))) _Float16 v16h;
typedef __attribute__((ext_vector_type(8)))  float    v8f;

#define BLK 256

__device__ inline unsigned ballot32(bool p) {
#if __has_builtin(__builtin_amdgcn_ballot_w32)
    return __builtin_amdgcn_ballot_w32(p);
#else
    return (unsigned)__ballot(p);
#endif
}

// ---------------------------------------------------------------------------
// K0: per-block count of segment boundaries (flag = p==0 || idx[p]!=idx[p-1])
// ---------------------------------------------------------------------------
__global__ void k0_count_bounds(const int* __restrict__ idx, int n,
                                int* __restrict__ blkCnt) {
    const int p = blockIdx.x * BLK + threadIdx.x;
    if (p + 4096 < n) __builtin_prefetch(&idx[p + 4096], 0, 1); // global_prefetch_b8
    bool flag = false;
    if (p < n) flag = (p == 0) || (idx[p] != idx[p - 1]);
    const unsigned bal = ballot32(flag);
    __shared__ int wcnt[BLK / 32];
    const int lane = threadIdx.x & 31, wid = threadIdx.x >> 5;
    if (lane == 0) wcnt[wid] = __popc(bal);
    __syncthreads();
    if (threadIdx.x == 0) {
        int s = 0;
        for (int i = 0; i < BLK / 32; ++i) s += wcnt[i];
        blkCnt[blockIdx.x] = s;
    }
}

// ---------------------------------------------------------------------------
// K1: single-wave exclusive scan of block counts via WMMA matmul-scan.
// Chunk of 256 counts -> X (16x16, row-major). P = X * U with U upper-tri
// ones gives per-row inclusive scans in ONE v_wmma_f32_16x16x32_f16
// (K columns 16..31 zero-padded). Counts <= 256 are exact in f16; all sums
// <= n_seg <= 50k are exact in f32.
// A layout (16-bit A 16x32): lane L (m=L%16, h=L/16), VGPR v in 0..3 holds
// halfs K = {2v+8h, 2v+1+8h}; VGPRs 4..7 hold K=16..31 (zeros here).
// B layout (16-bit B 32x16): lane n=L%16; VGPR v in 0..3 holds rows {2v+8h, 2v+1+8h};
// VGPRs 4..7 hold rows 16..31 (the zero padding).
// C/D layout: acc[v] at lane L = P[v + 8h][L%16].
// ---------------------------------------------------------------------------
__global__ void k1_scan_block_counts(const int* __restrict__ blkCnt,
                                     int* __restrict__ blkOff,
                                     int* __restrict__ nsegPtr, int nb) {
    __shared__ float xs[256];
    __shared__ float rts[16];
    const int lane = threadIdx.x;      // 32 threads, one wave, EXEC all ones
    const int m = lane & 15, h = lane >> 4;
    int carry = 0;
    const int nChunks = (nb + 255) / 256;
    for (int ch = 0; ch < nChunks; ++ch) {
        const int base = ch * 256;
        for (int i = lane; i < 256; i += 32) {
            const int g = base + i;
            xs[i] = (g < nb) ? (float)blkCnt[g] : 0.0f;
        }
        __syncthreads();

        v16h a = {};
#pragma unroll
        for (int v = 0; v < 4; ++v) {
            const int k0 = 2 * v + 8 * h;
            a[2 * v]     = (_Float16)xs[m * 16 + k0];
            a[2 * v + 1] = (_Float16)xs[m * 16 + k0 + 1];
        }
        v16h b = {};
        {
#pragma unroll
            for (int v = 0; v < 4; ++v) {
                b[2 * v]     = (_Float16)((2 * v + 8 * h     <= m) ? 1.0f : 0.0f);
                b[2 * v + 1] = (_Float16)((2 * v + 8 * h + 1 <= m) ? 1.0f : 0.0f);
            }
        }
        v8f acc = {};
        acc = __builtin_amdgcn_wmma_f32_16x16x32_f16(
            false, a, false, b, (short)0, acc, false, false);

        // publish row totals P[M][15]
        if (m == 15) {
#pragma unroll
            for (int v = 0; v < 8; ++v) rts[v + 8 * h] = acc[v];
        }
        __syncthreads();

#pragma unroll
        for (int v = 0; v < 8; ++v) {
            const int M = v + 8 * h;
            float ro = 0.0f;
            for (int i = 0; i < M; ++i) ro += rts[i];
            const int e = M * 16 + m;
            const int g = base + e;
            if (g < nb) {
                const float incl = ro + acc[v];                 // inclusive scan
                blkOff[g] = carry + (int)(incl + 0.5f) - (int)(xs[e] + 0.5f);
            }
        }
        float tot = 0.0f;
        for (int i = 0; i < 16; ++i) tot += rts[i];
        carry += (int)(tot + 0.5f);
        __syncthreads();
    }
    if (lane == 0) nsegPtr[0] = carry;   // n_seg
}

// ---------------------------------------------------------------------------
// K2: scatter segment start positions using block offset + ballot rank
// ---------------------------------------------------------------------------
__global__ void k2_write_seg_starts(const int* __restrict__ idx, int n,
                                    const int* __restrict__ blkOff,
                                    const int* __restrict__ nsegPtr,
                                    int* __restrict__ segStart) {
    const int p = blockIdx.x * BLK + threadIdx.x;
    const bool flag = (p < n) && ((p == 0) || (idx[p] != idx[p - 1]));
    const unsigned bal = ballot32(flag);
    const int lane = threadIdx.x & 31, wid = threadIdx.x >> 5;
    __shared__ int wcnt[BLK / 32];
    if (lane == 0) wcnt[wid] = __popc(bal);
    __syncthreads();
    int waveOff = 0;
    for (int i = 0; i < wid; ++i) waveOff += wcnt[i];
    if (flag) {
        const int rank = waveOff + __popc(bal & ((1u << lane) - 1u));
        segStart[blkOff[blockIdx.x] + rank] = p;
    }
    if (p == 0) segStart[nsegPtr[0]] = n;   // sentinel: end of last segment
}

// ---------------------------------------------------------------------------
// K3: single-block integer scan of tri counts C(c,2)  (total ~24.8M > 2^24,
// so this must stay integer — no f16/f32 matmul-scan here)
// ---------------------------------------------------------------------------
__global__ void k3_scan_tri(const int* __restrict__ segStart,
                            const int* __restrict__ nsegPtr,
                            int* __restrict__ triOff) {
    __shared__ int lds[BLK];
    __shared__ int sCarry;
    const int t = threadIdx.x;
    const int nseg = nsegPtr[0];
    if (t == 0) sCarry = 0;
    __syncthreads();
    const int nChunks = (nseg + BLK - 1) / BLK;
    for (int ch = 0; ch < nChunks; ++ch) {
        const int s = ch * BLK + t;
        int tri = 0;
        if (s < nseg) {
            const int c = segStart[s + 1] - segStart[s];
            tri = (c * (c - 1)) >> 1;
        }
        lds[t] = tri;
        __syncthreads();
        for (int d = 1; d < BLK; d <<= 1) {       // Hillis-Steele inclusive
            const int v = (t >= d) ? lds[t - d] : 0;
            __syncthreads();
            lds[t] += v;
            __syncthreads();
        }
        const int incl = lds[t];
        const int carry = sCarry;
        if (s < nseg) triOff[s] = carry + incl - tri;   // exclusive
        __syncthreads();
        if (t == BLK - 1) sCarry = carry + lds[BLK - 1];
        __syncthreads();
    }
    if (t == 0) triOff[nseg] = sCarry;   // == T
}

// ---------------------------------------------------------------------------
// K4: one wave per segment, flat coalesced emission with the same fp32
// triangular decode the reference uses. Streaming non-temporal stores:
// 3T*4B ~ 298MB output, bigger than the 192MB L2 and never reused.
// ---------------------------------------------------------------------------
__global__ void k4_emit_triples(const int* __restrict__ segStart,
                                const int* __restrict__ triOff,
                                const int* __restrict__ nsegPtr,
                                int* __restrict__ out, int T, int totalWaves) {
    const int gwave = (int)((blockIdx.x * blockDim.x + threadIdx.x) >> 5);
    const int lane = threadIdx.x & 31;
    const int nseg = nsegPtr[0];
    int* __restrict__ outI = out;
    int* __restrict__ outJ = out + T;
    int* __restrict__ outK = out + 2 * (size_t)T;
    for (int s = gwave; s < nseg; s += totalWaves) {
        const int base = segStart[s];
        const int c = segStart[s + 1] - base;
        const int M = (c * (c - 1)) >> 1;
        const int off = triOff[s];
        for (int r = lane; r < M; r += 32) {
            const int rr = M - 1 - r;
            const int a_rev =
                (int)floorf((sqrtf(8.0f * (float)rr + 1.0f) - 1.0f) * 0.5f);
            const int a = c - 2 - a_rev;
            const int b = c - 1 - (rr - ((a_rev * (a_rev + 1)) >> 1));
            __builtin_nontemporal_store(s, &outI[off + r]);
            __builtin_nontemporal_store(base + a, &outJ[off + r]);
            __builtin_nontemporal_store(base + b, &outK[off + r]);
        }
    }
}

// ---------------------------------------------------------------------------
extern "C" void kernel_launch(void* const* d_in, const int* in_sizes, int n_in,
                              void* d_out, int out_size, void* d_ws, size_t ws_size,
                              hipStream_t stream) {
    const int* idx = (const int*)d_in[0];
    const int n = in_sizes[0];
    int* out = (int*)d_out;          // (idx_i_triples | idx_j_triples | idx_k_triples)
    const int T = out_size / 3;

    const int nb = (n + BLK - 1) / BLK;
    int* ws = (int*)d_ws;
    int* blkCnt   = ws;                      // nb
    int* blkOff   = ws + nb;                 // nb
    int* nsegPtr  = ws + 2 * (size_t)nb;     // 1
    int* segStart = nsegPtr + 1;             // up to n+1
    int* triOff   = segStart + (size_t)n + 1;// up to n+1
    (void)ws_size; (void)n_in;

    k0_count_bounds<<<nb, BLK, 0, stream>>>(idx, n, blkCnt);
    k1_scan_block_counts<<<1, 32, 0, stream>>>(blkCnt, blkOff, nsegPtr, nb);
    k2_write_seg_starts<<<nb, BLK, 0, stream>>>(idx, n, blkOff, nsegPtr, segStart);
    k3_scan_tri<<<1, BLK, 0, stream>>>(segStart, nsegPtr, triOff);
    const int emitBlocks = 4096;
    k4_emit_triples<<<emitBlocks, BLK, 0, stream>>>(segStart, triOff, nsegPtr,
                                                    out, T, emitBlocks * (BLK / 32));
}
